// AttentionBlock_85005992723066
// MI455X (gfx1250) — matrix operand order repaired and checked by reading, not compiled, not run on hardware
//
#include <hip/hip_runtime.h>
#include <hip/hip_bf16.h>
#include <math.h>

// ---------------------------------------------------------------------------
// AttentionBlock (GroupNorm + spatial self-attention + proj + residual)
// B=8, C=256, H=W=64 -> N=4096 tokens, single head, d=256.
// ~155 GFLOP, dominated by N^2*C attention matmuls -> WMMA bf16 (f32 accum),
// flash-attention so the 4096x4096 score matrix is never materialized.
// K/V blocks are staged LDS-direct with GLOBAL_LOAD_ASYNC_TO_LDS_B128 and
// double-buffered via ASYNCcnt so DMA overlaps the WMMA phase.
// Workspace use: ~84.5 MB (h/Q/K/Vt/O bf16 token matrices + bf16 weights).
// ---------------------------------------------------------------------------

typedef __attribute__((ext_vector_type(16))) __bf16 v16bf;
typedef __attribute__((ext_vector_type(8)))  __bf16 v8bf;
typedef __attribute__((ext_vector_type(8)))  float  v8f;

#define kB   8
#define kC   256
#define kHW  4096
#define kG   8
#define kCPG 32      // channels per group
#define QB   64      // query rows per workgroup (flash)
#define KBL  32      // key block per iteration (flash)

__device__ __forceinline__ __bf16 f2bf(float f) {
  union { float f; unsigned u; } a; a.f = f;
  unsigned u = a.u + 0x7fffu + ((a.u >> 16) & 1u);   // round-to-nearest-even
  union { unsigned short s; __bf16 b; } r; r.s = (unsigned short)(u >> 16);
  return r.b;
}

__device__ __forceinline__ v8f vzero8() {
  v8f z;
#pragma unroll
  for (int i = 0; i < 8; ++i) z[i] = 0.f;
  return z;
}

// A fragment (16x32 bf16, MxK) from row-major src (pitch elements per row).
// ISA 7.12.2: lane m(0..15)=row m holds K=0..7,16..23; lanes 16..31 hold
// K=8..15,24..31 for the same rows.
__device__ __forceinline__ v16bf load_a_frag(const __bf16* base, int pitch, int lane) {
  const int m  = lane & 15;
  const int kb = (lane >> 4) << 3;               // 0 or 8
  const __bf16* p = base + m * pitch + kb;
  v8bf lo = *(const v8bf*)(p);                   // K = kb .. kb+7
  v8bf hi = *(const v8bf*)(p + 16);              // K = kb+16 .. kb+23
  return __builtin_shufflevector(lo, hi, 0,1,2,3,4,5,6,7,8,9,10,11,12,13,14,15);
}

// B fragment (32x16 bf16, KxN) from N-major storage bt[n][k] (pitch elems).
// lane n(0..15)=col n holds K=0..7,16..23; lanes 16..31 hold K=8..15,24..31.
__device__ __forceinline__ v16bf load_b_frag(const __bf16* bt, int pitch, int lane) {
  const int n  = lane & 15;
  const int kb = (lane >> 4) << 3;               // 0 or 8
  const __bf16* p = bt + n * pitch + kb;
  v8bf lo = *(const v8bf*)(p);
  v8bf hi = *(const v8bf*)(p + 16);
  return __builtin_shufflevector(lo, hi, 0,1,2,3,4,5,6,7,8,9,10,11,12,13,14,15);
}

__device__ __forceinline__ v8f wmma_bf16(v16bf a, v16bf b, v8f c) {
  return __builtin_amdgcn_wmma_f32_16x16x32_bf16(false, a, false, b, (short)0, c,
                                                 false, false);
}

// Async global->LDS copy, 16 bytes per lane (ASYNCcnt-tracked, no VGPR data).
// Generic LDS pointer low 32 bits == hardware LDS byte address (aperture rule:
// the LDS address is the generic address truncated to 32 bits).
__device__ __forceinline__ void async_copy_b128(void* lds_dst, const void* gsrc) {
  asm volatile("global_load_async_to_lds_b128 %0, %1, off"
               :: "v"((unsigned)(size_t)lds_dst),
                  "v"((unsigned long long)(size_t)gsrc)
               : "memory");
}

// -------------------------- GroupNorm statistics ---------------------------
__global__ __launch_bounds__(256) void gn_stats_kernel(const float* __restrict__ x,
                                                       float* __restrict__ stats) {
  const int bg = blockIdx.x;                     // b*kG + g ; group data contiguous
  const float* p = x + (size_t)bg * (kCPG * kHW);
  float s = 0.f, ss = 0.f;
  for (int i = threadIdx.x * 4; i < kCPG * kHW; i += 256 * 4) {
    float4 v = *(const float4*)(p + i);
    s  += v.x + v.y + v.z + v.w;
    ss += v.x*v.x + v.y*v.y + v.z*v.z + v.w*v.w;
  }
  __shared__ float rs[256], rq[256];
  rs[threadIdx.x] = s; rq[threadIdx.x] = ss;
  __syncthreads();
  for (int off = 128; off > 0; off >>= 1) {
    if (threadIdx.x < off) {
      rs[threadIdx.x] += rs[threadIdx.x + off];
      rq[threadIdx.x] += rq[threadIdx.x + off];
    }
    __syncthreads();
  }
  if (threadIdx.x == 0) {
    const float inv = 1.f / (float)(kCPG * kHW);
    const float mean = rs[0] * inv;
    const float var  = rq[0] * inv - mean * mean;
    stats[2*bg]     = mean;
    stats[2*bg + 1] = rsqrtf(var + 1e-5f);
  }
}

// ------------- normalize + transpose [B,C,HW] -> bf16 [B,HW,C] -------------
__global__ __launch_bounds__(256) void gn_apply_kernel(const float* __restrict__ x,
                                                       const float* __restrict__ stats,
                                                       __bf16* __restrict__ h) {
  __shared__ __bf16 tile[32][33];
  const int b  = blockIdx.z;
  const int c0 = blockIdx.y * 32;
  const int p0 = blockIdx.x * 32;
  const int tx = threadIdx.x & 31;
  const int ty = threadIdx.x >> 5;               // 0..7
#pragma unroll
  for (int i = 0; i < 4; ++i) {
    const int c = c0 + ty + i*8;
    const float mean = stats[2*(b*kG + (c >> 5))];
    const float rstd = stats[2*(b*kG + (c >> 5)) + 1];
    const float v = x[((size_t)b*kC + c)*kHW + p0 + tx];
    tile[ty + i*8][tx] = f2bf((v - mean) * rstd);
  }
  __syncthreads();
#pragma unroll
  for (int i = 0; i < 4; ++i) {
    const int pp = p0 + ty + i*8;
    h[((size_t)b*kHW + pp)*kC + c0 + tx] = tile[tx][ty + i*8];
  }
}

// -------------------------- weights fp32 -> bf16 ---------------------------
__global__ __launch_bounds__(256) void cvt_w_kernel(
    const float* __restrict__ w0, const float* __restrict__ w1,
    const float* __restrict__ w2, const float* __restrict__ w3,
    __bf16* __restrict__ o0, __bf16* __restrict__ o1,
    __bf16* __restrict__ o2, __bf16* __restrict__ o3) {
  const int m = blockIdx.y;
  const float* w = (m == 0) ? w0 : (m == 1) ? w1 : (m == 2) ? w2 : w3;
  __bf16*      o = (m == 0) ? o0 : (m == 1) ? o1 : (m == 2) ? o2 : o3;
  const int i = blockIdx.x * 256 + threadIdx.x;
  o[i] = f2bf(w[i]);
}

// --------------- QKV projections: Q,K [N,C] and V transposed ---------------
__global__ __launch_bounds__(128) void qkv_gemm_kernel(
    const __bf16* __restrict__ h,
    const __bf16* __restrict__ wq, const __bf16* __restrict__ wk,
    const __bf16* __restrict__ wv,
    const float* __restrict__ bq, const float* __restrict__ bk,
    const float* __restrict__ bv,
    __bf16* __restrict__ Q, __bf16* __restrict__ K, __bf16* __restrict__ Vt) {
  const int lane = threadIdx.x & 31;
  const int wave = threadIdx.x >> 5;
  const int b  = blockIdx.x / (kHW / 16);
  const int m0 = (blockIdx.x % (kHW / 16)) * 16;
  const int mat = blockIdx.y;                    // 0=Q 1=K 2=V
  const __bf16* W    = (mat == 0) ? wq : (mat == 1) ? wk : wv;
  const float*  bias = (mat == 0) ? bq : (mat == 1) ? bk : bv;

  const __bf16* A = h + ((size_t)b*kHW + m0) * kC;
  v16bf af[8];
#pragma unroll
  for (int c = 0; c < 8; ++c) af[c] = load_a_frag(A + c*32, kC, lane);

  const int col = lane & 15;
  const int rb  = (lane >> 4) * 8;
#pragma unroll
  for (int t = 0; t < 4; ++t) {
    const int o0 = wave*64 + t*16;               // out = h @ W^T : B-frag = W rows
    v8f acc = vzero8();
#pragma unroll
    for (int c = 0; c < 8; ++c)
      acc = wmma_bf16(af[c], load_b_frag(W + (size_t)o0*kC + c*32, kC, lane), acc);
    const float bb = bias[o0 + col];
    if (mat == 2) {                              // V stored transposed [C, N]
      __bf16* dst = Vt + ((size_t)b*kC + (o0 + col)) * kHW + m0;
#pragma unroll
      for (int r = 0; r < 8; ++r) dst[rb + r] = f2bf(acc[r] + bb);
    } else {
      __bf16* dst = ((mat == 0) ? Q : K) + ((size_t)b*kHW + m0) * kC;
#pragma unroll
      for (int r = 0; r < 8; ++r)
        dst[(size_t)(rb + r)*kC + o0 + col] = f2bf(acc[r] + bb);
    }
  }
}

// ------------------------------ flash attention ----------------------------
// 4 waves per block; wave owns 16 query rows, Q-frags register resident.
// Per key block (32 keys): 16 WMMAs for S, online softmax via LDS, 16 WMMAs
// for O += P@V.  K/Vt blocks are double-buffered in LDS and filled with
// GLOBAL_LOAD_ASYNC_TO_LDS_B128: each wave issues 16 async b128s per buffer,
// waits ASYNCcnt<=16 (in-order completion => current buffer done), then a
// workgroup barrier publishes the buffer to all waves.
__global__ __launch_bounds__(128) void flash_attn_kernel(
    const __bf16* __restrict__ Q, const __bf16* __restrict__ K,
    const __bf16* __restrict__ Vt, __bf16* __restrict__ O) {
  __shared__ __align__(16) __bf16 sK [2][KBL * kC];   // 2 x 16 KB
  __shared__ __align__(16) __bf16 sVt[2][kC * KBL];   // 2 x 16 KB
  __shared__ __align__(16) float  sS[4][16 * 32];     // per-wave score tile
  __shared__ __align__(16) __bf16 sP[4][16 * 32];     // per-wave P tile
  __shared__ float sC[4][16];
  __shared__ float sL[4][16];

  const int tid  = threadIdx.x;
  const int lane = tid & 31;
  const int wave = tid >> 5;
  const int b  = blockIdx.x / (kHW / QB);
  const int q0 = (blockIdx.x % (kHW / QB)) * QB + wave * 16;

  const __bf16* Qb = Q  + ((size_t)b*kHW + q0) * kC;
  const __bf16* Kb = K  + (size_t)b*kHW*kC;
  const __bf16* Vb = Vt + (size_t)b*(size_t)kC*kHW;

  // stage one K/Vt block into LDS buffer `buf` (16 async b128 per thread)
  auto stage_async = [&](int buf, int kb) {
    const __bf16* ksrc = Kb + (size_t)kb * kC;     // 32 rows x 256 ch: contiguous
#pragma unroll
    for (int i = 0; i < 8; ++i) {
      const int idx = tid*8 + i*1024;
      async_copy_b128(&sK[buf][idx], ksrc + idx);
    }
#pragma unroll
    for (int rr = 0; rr < 2; ++rr) {               // Vt block: 256 rows x 32 keys
      const int ch = tid + rr*128;
      const __bf16* vsrc = Vb + (size_t)ch*kHW + kb;
      __bf16* vdst = &sVt[buf][ch*KBL];
#pragma unroll
      for (int j = 0; j < 4; ++j)
        async_copy_b128(vdst + j*8, vsrc + j*8);
    }
  };

  v16bf qf[8];
#pragma unroll
  for (int c = 0; c < 8; ++c) qf[c] = load_a_frag(Qb + c*32, kC, lane);

  v8f acc[16];                                    // O tile: 16 x 256 fp32
#pragma unroll
  for (int t = 0; t < 16; ++t) acc[t] = vzero8();

  float mrow = -__builtin_inff(), lrow = 0.f;     // lanes 0..15 own row=lane
  const float scale = 0.0625f;                    // C^-0.5 = 1/16
  const int col = lane & 15;
  const int rb  = (lane >> 4) * 8;

  stage_async(0, 0);                              // prologue

  for (int kb = 0; kb < kHW; kb += KBL) {
    const int cur = (kb / KBL) & 1;
    if (kb + KBL < kHW) {
      stage_async(cur ^ 1, kb + KBL);             // prefetch next block
      asm volatile("s_wait_asynccnt 0x10" ::: "memory");  // cur block landed
    } else {
      asm volatile("s_wait_asynccnt 0x0" ::: "memory");
    }
    __syncthreads();                              // publish cur buffer

    v8f s0 = vzero8(), s1 = vzero8();             // S = Q*K^T (16 x 32)
    const __bf16* sk = &sK[cur][0];
#pragma unroll
    for (int c = 0; c < 8; ++c) {
      s0 = wmma_bf16(qf[c], load_b_frag(sk + c*32, kC, lane), s0);
      s1 = wmma_bf16(qf[c], load_b_frag(sk + 16*kC + c*32, kC, lane), s1);
    }
    {
      float* Sw = &sS[wave][0];                   // D layout: lane=col, 8 rows
#pragma unroll
      for (int r = 0; r < 8; ++r) {
        Sw[(rb + r)*32 + col]      = s0[r] * scale;
        Sw[(rb + r)*32 + col + 16] = s1[r] * scale;
      }
    }
    __syncthreads();

    if (lane < 16) {                              // online softmax, row = lane
      const float* Sw = &sS[wave][0] + lane*32;
      float mx = mrow;
#pragma unroll
      for (int j = 0; j < 32; ++j) mx = fmaxf(mx, Sw[j]);
      const float corr = __expf(mrow - mx);
      float sum = 0.f;
      __bf16* Pw = &sP[wave][0] + lane*32;
#pragma unroll
      for (int j = 0; j < 32; ++j) {
        const float e = __expf(Sw[j] - mx);
        sum += e;
        Pw[j] = f2bf(e);
      }
      lrow = lrow * corr + sum;
      mrow = mx;
      sC[wave][lane] = corr;
    }
    __syncthreads();

    float cf[8];                                  // rescale O by row correction
#pragma unroll
    for (int r = 0; r < 8; ++r) cf[r] = sC[wave][rb + r];
#pragma unroll
    for (int t = 0; t < 16; ++t)
#pragma unroll
      for (int r = 0; r < 8; ++r) acc[t][r] *= cf[r];

    const v16bf pf = load_a_frag(&sP[wave][0], 32, lane);
    const __bf16* sv = &sVt[cur][0];
#pragma unroll
    for (int t = 0; t < 16; ++t)                  // O += P @ V
      acc[t] = wmma_bf16(pf, load_b_frag(sv + t*16*KBL, KBL, lane), acc[t]);

    __syncthreads();                              // reads of cur done -> reusable
  }

  if (lane < 16) sL[wave][lane] = 1.f / lrow;
  __syncthreads();
  float li[8];
#pragma unroll
  for (int r = 0; r < 8; ++r) li[r] = sL[wave][rb + r];

  __bf16* Ob = O + ((size_t)b*kHW + q0) * kC;
#pragma unroll
  for (int t = 0; t < 16; ++t) {
    const int ch = t*16 + col;
#pragma unroll
    for (int r = 0; r < 8; ++r)
      Ob[(size_t)(rb + r)*kC + ch] = f2bf(acc[t][r] * li[r]);
  }
}

// ----------------- output projection + bias + residual ---------------------
__global__ __launch_bounds__(128) void out_proj_kernel(
    const __bf16* __restrict__ Oa, const __bf16* __restrict__ wo,
    const float* __restrict__ bo, const float* __restrict__ x,
    float* __restrict__ out) {
  const int lane = threadIdx.x & 31;
  const int wave = threadIdx.x >> 5;
  const int b  = blockIdx.x / (kHW / 16);
  const int m0 = (blockIdx.x % (kHW / 16)) * 16;

  const __bf16* A = Oa + ((size_t)b*kHW + m0) * kC;
  v16bf af[8];
#pragma unroll
  for (int c = 0; c < 8; ++c) af[c] = load_a_frag(A + c*32, kC, lane);

  const int col = lane & 15;
  const int rb  = (lane >> 4) * 8;
#pragma unroll
  for (int t = 0; t < 4; ++t) {
    const int o0 = wave*64 + t*16;
    v8f accv = vzero8();
#pragma unroll
    for (int c = 0; c < 8; ++c)
      accv = wmma_bf16(af[c], load_b_frag(wo + (size_t)o0*kC + c*32, kC, lane), accv);
    const float bb = bo[o0 + col];
    // D tile: lane = out-channel, 8 consecutive tokens -> contiguous in [B,C,HW]
    const size_t base = ((size_t)b*kC + (o0 + col)) * kHW + m0 + rb;
#pragma unroll
    for (int r = 0; r < 8; ++r)
      out[base + r] = x[base + r] + accv[r] + bb;
  }
}

// ---------------------------------------------------------------------------
extern "C" void kernel_launch(void* const* d_in, const int* in_sizes, int n_in,
                              void* d_out, int out_size, void* d_ws, size_t ws_size,
                              hipStream_t stream) {
  (void)in_sizes; (void)n_in; (void)out_size; (void)ws_size;
  const float* x  = (const float*)d_in[0];
  const float* wq = (const float*)d_in[1];
  const float* bq = (const float*)d_in[2];
  const float* wk = (const float*)d_in[3];
  const float* bk = (const float*)d_in[4];
  const float* wv = (const float*)d_in[5];
  const float* bv = (const float*)d_in[6];
  const float* wo = (const float*)d_in[7];
  const float* bo = (const float*)d_in[8];
  float* out = (float*)d_out;

  // workspace carve-up (~84.5 MB total)
  char* ws = (char*)d_ws;
  float* stats = (float*)ws;                      // 64 * {mean, rstd}
  size_t off = 1024;
  const size_t tok_mat = (size_t)kB * kHW * kC;   // 8Mi elements
  __bf16* h   = (__bf16*)(ws + off); off += tok_mat * 2;   // normalized input
  __bf16* Qb  = (__bf16*)(ws + off); off += tok_mat * 2;
  __bf16* Kb  = (__bf16*)(ws + off); off += tok_mat * 2;
  __bf16* Vtb = (__bf16*)(ws + off); off += tok_mat * 2;   // V transposed [C,N]
  __bf16* Ob  = (__bf16*)(ws + off); off += tok_mat * 2;   // attention output
  __bf16* wqb = (__bf16*)(ws + off); off += (size_t)kC * kC * 2;
  __bf16* wkb = (__bf16*)(ws + off); off += (size_t)kC * kC * 2;
  __bf16* wvb = (__bf16*)(ws + off); off += (size_t)kC * kC * 2;
  __bf16* wob = (__bf16*)(ws + off); off += (size_t)kC * kC * 2;

  gn_stats_kernel<<<kB * kG, 256, 0, stream>>>(x, stats);
  gn_apply_kernel<<<dim3(kHW / 32, kC / 32, kB), 256, 0, stream>>>(x, stats, h);
  cvt_w_kernel<<<dim3(kC * kC / 256, 4), 256, 0, stream>>>(wq, wk, wv, wo,
                                                           wqb, wkb, wvb, wob);
  qkv_gemm_kernel<<<dim3(kB * (kHW / 16), 3), 128, 0, stream>>>(
      h, wqb, wkb, wvb, bq, bk, bv, Qb, Kb, Vtb);
  flash_attn_kernel<<<kB * (kHW / QB), 128, 0, stream>>>(Qb, Kb, Vtb, Ob);
  out_proj_kernel<<<kB * (kHW / 16), 128, 0, stream>>>(Ob, wob, bo, x, out);
}
